// LSTM_Layer_81046032875700
// MI455X (gfx1250) — compile-verified, matrix operand order checked by reading, not run on hardware
//
#include <hip/hip_runtime.h>
#include <hip/hip_bf16.h>

// ---------------------------------------------------------------------------
// LSTM layer, persistent-kernel formulation for MI455X (gfx1250, wave32).
//
//   z_t = x_t @ Wx + h_{t-1} @ Wh + b            (64x512 @ 512x2048, twice)
//   i,f,g,o gates -> c_t, h_t ;  out[t] = h_t    (out: [512][64][512] f32)
//
// 128 workgroups (4 row-tiles x 32 unit-tiles), 4 wave32 each, one wave per
// gate. Weight column slices staged once into LDS as bf16. Per step, each
// wave runs FOUR independent WMMA chains of 8 (x/h part x K-half) to cut the
// dependent-latency of the serial step. A/h tiles are staged into LDS with
// gfx1250 global_load_async_to_lds_b128. Timesteps separated by a two-level
// agent-scope atomic barrier (4 leaf lines + root + phase).
// ---------------------------------------------------------------------------

#define UNITS   512
#define INPUT   512
#define BATCH   64
#define SEQLEN  512
#define GSTRIDE 2048            // 4*UNITS
#define TM      16
#define TN      16
#define NROWT   (BATCH / TM)    // 4
#define NCOLT   (UNITS / TN)    // 32
#define NWG     (NROWT * NCOLT) // 128
#define NTHR    128             // 4 wave32
#define NLEAF   4
#define LEAFCNT (NWG / NLEAF)   // 32 arrivals per leaf

// workspace layout (bytes)
#define ST_OFF    0                                   // barrier state (512 B)
#define CBUF_OFF  512                                 // f32 [64][512]
#define HBUF_OFF  (CBUF_OFF + BATCH * UNITS * 4)      // bf16 [2][64][512]
#define XBF_OFF   (HBUF_OFF + 2 * BATCH * UNITS * 2)  // bf16 [64][512][512] (optional)
#define XBF_BYTES ((size_t)BATCH * SEQLEN * INPUT * 2)
#define WS_NEED_ASYNC_X ((size_t)XBF_OFF + XBF_BYTES)

typedef __attribute__((ext_vector_type(16))) __bf16 v16bf;
typedef __attribute__((ext_vector_type(8)))  float  v8f;

struct BF16Frag { uint4 lo, hi; };   // 32 bytes == 16 bf16

// Per-lane fragment load for 16-bit WMMA operands (ISA 7.12.2 layout):
// lane L (half = L>>4) needs elements [off, off+8) and [off+16, off+24).
__device__ __forceinline__ v16bf load_frag(const __hip_bfloat16* row, int off) {
    BF16Frag f;
    f.lo = *(const uint4*)(row + off);
    f.hi = *(const uint4*)(row + off + 16);
    return __builtin_bit_cast(v16bf, f);
}

__device__ __forceinline__ float fast_tanh(float x) {
#if __has_builtin(__builtin_amdgcn_tanhf)
    return __builtin_amdgcn_tanhf(x);          // v_tanh_f32 (gfx1250 trans op)
#else
    float e = __expf(2.0f * x);                // branch-free fallback
    return (e - 1.0f) / (e + 1.0f);
#endif
}
__device__ __forceinline__ float fast_sigmoid(float x) {
    return 0.5f * fast_tanh(0.5f * x) + 0.5f;
}

// gfx1250 async copy global -> LDS (16B per lane), tracked by ASYNCcnt.
__device__ __forceinline__ void async_load_b128(unsigned lds_off, const void* gptr) {
    asm volatile("global_load_async_to_lds_b128 %0, %1, off"
                 :: "v"(lds_off), "v"(gptr) : "memory");
}
__device__ __forceinline__ void async_wait0() {
    asm volatile("s_wait_asynccnt 0" ::: "memory");
}
// generic LDS pointer -> 32-bit LDS byte offset (hw truncates upper bits)
__device__ __forceinline__ unsigned lds_off_of(const void* p) {
    return (unsigned)(uintptr_t)p;
}

// two-level grid barrier: 4 leaf counters (64B apart) -> root -> phase
__device__ __forceinline__ void grid_barrier(unsigned* st, int t, int wg) {
    __threadfence();
    __syncthreads();
    if (threadIdx.x == 0) {
        unsigned* leaf  = &st[(wg & (NLEAF - 1)) * 16];
        unsigned* root  = &st[NLEAF * 16];
        unsigned* phase = &st[(NLEAF + 1) * 16];
        unsigned a = __hip_atomic_fetch_add(leaf, 1u, __ATOMIC_ACQ_REL,
                                            __HIP_MEMORY_SCOPE_AGENT);
        if (a == LEAFCNT - 1) {
            __hip_atomic_store(leaf, 0u, __ATOMIC_RELAXED, __HIP_MEMORY_SCOPE_AGENT);
            unsigned r = __hip_atomic_fetch_add(root, 1u, __ATOMIC_ACQ_REL,
                                                __HIP_MEMORY_SCOPE_AGENT);
            if (r == NLEAF - 1) {
                __hip_atomic_store(root, 0u, __ATOMIC_RELAXED, __HIP_MEMORY_SCOPE_AGENT);
                __hip_atomic_fetch_add(phase, 1u, __ATOMIC_RELEASE,
                                       __HIP_MEMORY_SCOPE_AGENT);
            }
        }
        while (__hip_atomic_load(phase, __ATOMIC_ACQUIRE,
                                 __HIP_MEMORY_SCOPE_AGENT) < (unsigned)(t + 1)) {
            __builtin_amdgcn_s_sleep(2);
        }
    }
    __syncthreads();
}

__global__ void lstm_init(const float* __restrict__ h0,
                          const float* __restrict__ c0,
                          float* __restrict__ cbuf,
                          __hip_bfloat16* __restrict__ hbuf,
                          unsigned* __restrict__ st) {
    int i = blockIdx.x * blockDim.x + threadIdx.x;
    if (i < 128) st[i] = 0u;                 // whole barrier block
    if (i < BATCH * UNITS) {
        cbuf[i] = c0[i];
        hbuf[i] = __float2bfloat16(h0[i]);   // ping buffer 0 = h_{-1}
    }
}

__global__ void convert_x(const float* __restrict__ x,
                          __hip_bfloat16* __restrict__ xbf) {
    int i = blockIdx.x * blockDim.x + threadIdx.x;  // grid sized exactly
    xbf[i] = __float2bfloat16(x[i]);
}

__global__ __launch_bounds__(NTHR, 1)
void lstm_persistent(const float* __restrict__ x,       // [BATCH][SEQ][INPUT]
                     const __hip_bfloat16* __restrict__ xbf, // optional bf16 copy
                     int use_xbf,
                     const float* __restrict__ Wx,      // [INPUT][4*UNITS]
                     const float* __restrict__ Wh,      // [UNITS][4*UNITS]
                     const float* __restrict__ bias,    // [4*UNITS]
                     float* __restrict__ out,           // [SEQ][BATCH][UNITS]
                     float* __restrict__ cbuf,          // [BATCH][UNITS]
                     __hip_bfloat16* __restrict__ hbuf, // [2][BATCH][UNITS]
                     unsigned* __restrict__ st) {
    // LDS: 64+64+16+16+4 KB = ~164 KB (<= 320 KB/WGP on CDNA5)
    __shared__ __align__(16) __hip_bfloat16 sWxT[64][INPUT]; // [g*16+n][k]
    __shared__ __align__(16) __hip_bfloat16 sWhT[64][UNITS];
    __shared__ __align__(16) __hip_bfloat16 sAx[TM][INPUT];  // x tile, bf16
    __shared__ __align__(16) __hip_bfloat16 sAh[TM][UNITS];  // h tile, bf16
    __shared__ __align__(16) float sZ[4][TM][TN];            // gate pre-acts

    const int tid      = threadIdx.x;
    const int wave     = tid >> 5;          // gate id: 0=i 1=f 2=g 3=o
    const int lane     = tid & 31;
    const int m_lane   = lane & 15;
    const int n_lane   = lane & 15;
    const int half     = lane >> 4;         // k-half selector for 16b frags
    const int wg       = blockIdx.x;
    const int rowt     = wg & (NROWT - 1);
    const int colt     = wg >> 2;
    const int rowbase  = rowt * TM;
    const int unitbase = colt * TN;

    // ---- stage transposed weight column slices (once) --------------------
    for (int idx = tid; idx < 64 * INPUT; idx += NTHR) {
        int c = idx >> 9;            // 0..63  (INPUT == 512)
        int k = idx & (INPUT - 1);
        int g = c >> 4;
        int n = c & 15;
        int col = g * UNITS + unitbase + n;
        sWxT[c][k] = __float2bfloat16(Wx[k * GSTRIDE + col]);
        sWhT[c][k] = __float2bfloat16(Wh[k * GSTRIDE + col]);
    }

    const float bval = bias[wave * UNITS + unitbase + n_lane];
    __syncthreads();

    for (int t = 0; t < SEQLEN; ++t) {
        const __hip_bfloat16* hprev = hbuf + (size_t)(t & 1) * BATCH * UNITS;
        __hip_bfloat16*       hnext = hbuf + (size_t)((t + 1) & 1) * BATCH * UNITS;

        // ---- stage A tiles for this step ---------------------------------
        if (use_xbf) {
            // async DMA path: 1024 x 16B chunks, 8 per thread
            #pragma unroll
            for (int i = 0; i < 8; ++i) {
                int idx = tid + i * NTHR;          // 0..1023
                int m   = idx >> 6;                // 64 chunks per 512-elem row
                int kc  = idx & 63;
                const __hip_bfloat16* g =
                    xbf + ((size_t)(rowbase + m) * SEQLEN + t) * INPUT + kc * 8;
                async_load_b128(lds_off_of(&sAx[m][kc * 8]), g);
                if (t + 1 < SEQLEN)
                    __builtin_prefetch(g + INPUT, 0, 1);   // next step's x
            }
        } else {
            for (int idx = tid; idx < TM * INPUT; idx += NTHR) {
                int m = idx >> 9;
                int k = idx & (INPUT - 1);
                const float* xp =
                    &x[((size_t)(rowbase + m) * SEQLEN + t) * INPUT + k];
                sAx[m][k] = __float2bfloat16(*xp);
                if (t + 1 < SEQLEN) __builtin_prefetch(xp + INPUT, 0, 1);
            }
        }
        // h tile: always async (hbuf is bf16 in workspace)
        #pragma unroll
        for (int i = 0; i < 8; ++i) {
            int idx = tid + i * NTHR;
            int m   = idx >> 6;
            int kc  = idx & 63;
            async_load_b128(lds_off_of(&sAh[m][kc * 8]),
                            hprev + (size_t)(rowbase + m) * UNITS + kc * 8);
        }
        async_wait0();
        __syncthreads();

        // ---- 4 independent WMMA chains of 8 (x/h part x K-half) ----------
        v8f a0, a1, a2, a3;
        #pragma unroll
        for (int i = 0; i < 8; ++i) { a0[i] = bval; a1[i] = 0.f; a2[i] = 0.f; a3[i] = 0.f; }

        const __hip_bfloat16* arow_x = &sAx[m_lane][0];
        const __hip_bfloat16* arow_h = &sAh[m_lane][0];
        const __hip_bfloat16* brow_x = &sWxT[wave * 16 + n_lane][0];
        const __hip_bfloat16* brow_h = &sWhT[wave * 16 + n_lane][0];

        #pragma unroll
        for (int kk = 0; kk < 8; ++kk) {
            const int off = kk * 32 + half * 8;   // K-half 0: elems [0,256)
            a0 = __builtin_amdgcn_wmma_f32_16x16x32_bf16(
                     false, load_frag(arow_x, off),       false,
                     load_frag(brow_x, off),       (short)0, a0, false, false);
            a1 = __builtin_amdgcn_wmma_f32_16x16x32_bf16(
                     false, load_frag(arow_x, off + 256), false,
                     load_frag(brow_x, off + 256), (short)0, a1, false, false);
            a2 = __builtin_amdgcn_wmma_f32_16x16x32_bf16(
                     false, load_frag(arow_h, off),       false,
                     load_frag(brow_h, off),       (short)0, a2, false, false);
            a3 = __builtin_amdgcn_wmma_f32_16x16x32_bf16(
                     false, load_frag(arow_h, off + 256), false,
                     load_frag(brow_h, off + 256), (short)0, a3, false, false);
        }
        v8f acc = (a0 + a1) + (a2 + a3);

        // ---- spill gate tile (C layout: VGPR i -> row i + half*8) --------
        #pragma unroll
        for (int i = 0; i < 8; ++i)
            sZ[wave][i + half * 8][n_lane] = acc[i];
        __syncthreads();

        // ---- elementwise gates: 2 of the 256 tile elements per thread ----
        #pragma unroll
        for (int j = 0; j < 2; ++j) {
            int idx = tid * 2 + j;
            int m = idx >> 4;
            int n = idx & 15;
            float ig = fast_sigmoid(sZ[0][m][n]);
            float fg = fast_sigmoid(sZ[1][m][n]);
            float gg = fast_tanh(sZ[2][m][n]);
            float og = fast_sigmoid(sZ[3][m][n]);
            int gidx = (rowbase + m) * UNITS + unitbase + n;
            float cn = fg * cbuf[gidx] + ig * gg;
            cbuf[gidx] = cn;
            float hn = og * fast_tanh(cn);
            out[(size_t)t * BATCH * UNITS + gidx] = hn;
            hnext[gidx] = __float2bfloat16(hn);
        }

        grid_barrier(st, t, wg);   // step t published before t+1 reads h
    }
}

extern "C" void kernel_launch(void* const* d_in, const int* in_sizes, int n_in,
                              void* d_out, int out_size, void* d_ws, size_t ws_size,
                              hipStream_t stream) {
    const float* x  = (const float*)d_in[0];
    const float* h0 = (const float*)d_in[1];
    const float* c0 = (const float*)d_in[2];
    const float* Wx = (const float*)d_in[3];
    const float* Wh = (const float*)d_in[4];
    const float* b  = (const float*)d_in[5];
    float* out = (float*)d_out;

    unsigned*       st   = (unsigned*)((char*)d_ws + ST_OFF);
    float*          cbuf = (float*)((char*)d_ws + CBUF_OFF);
    __hip_bfloat16* hbuf = (__hip_bfloat16*)((char*)d_ws + HBUF_OFF);
    __hip_bfloat16* xbf  = (__hip_bfloat16*)((char*)d_ws + XBF_OFF);

    const int use_xbf = (ws_size >= WS_NEED_ASYNC_X) ? 1 : 0;

    lstm_init<<<(BATCH * UNITS + 255) / 256, 256, 0, stream>>>(h0, c0, cbuf, hbuf, st);
    if (use_xbf) {
        const int n = BATCH * SEQLEN * INPUT;          // 16,777,216 (mult of 256)
        convert_x<<<n / 256, 256, 0, stream>>>(x, xbf);
    }
    lstm_persistent<<<NWG, NTHR, 0, stream>>>(x, xbf, use_xbf, Wx, Wh, b,
                                              out, cbuf, hbuf, st);
}
